// TemporalExtGCN_9208409883103
// MI455X (gfx1250) — matrix operand order repaired and checked by reading, not compiled, not run on hardware
//
#include <hip/hip_runtime.h>
#include <hip/hip_bf16.h>
#include <stdint.h>

// ---------------------------------------------------------------------------
// RGCN (R=1) + global_add_pool + FC, tuned for MI455X (gfx1250, wave32, WMMA,
// async global->LDS copies, double-buffered LDS pipeline)
// ---------------------------------------------------------------------------

typedef __attribute__((ext_vector_type(16))) __bf16 v16bf;
typedef __attribute__((ext_vector_type(8)))  float  v8f;

#define F_IN   500
#define KPAD   512      // pad K to a multiple of 32 for 16x16x32 bf16 WMMA
#define F_HID  1024
#define NGRAPH 64

union FragBF { v16bf v; uint4 u[2]; };

// -------- gfx1250 async global->LDS copy (ASYNCcnt path), with fallback ----
#if defined(__HIP_DEVICE_COMPILE__) && __has_builtin(__builtin_amdgcn_global_load_async_to_lds_b128)
#define HAVE_ASYNC_LDS 1
#else
#define HAVE_ASYNC_LDS 0
#endif

#if HAVE_ASYNC_LDS
// per clang diagnostic: params are pointers to 16-byte int vectors,
// global (AS1) source and LDS (AS3) destination
typedef int v4i_b128 __attribute__((vector_size(4 * sizeof(int))));
typedef __attribute__((address_space(1))) v4i_b128* gptr_b128;
typedef __attribute__((address_space(3))) v4i_b128* lptr_b128;
#endif

__device__ __forceinline__ void async_copy_b128(const void* g, void* l) {
#if HAVE_ASYNC_LDS
    __builtin_amdgcn_global_load_async_to_lds_b128(
        (gptr_b128)(uintptr_t)g, (lptr_b128)l, 0, 0);
#else
    *reinterpret_cast<uint4*>(l) = *reinterpret_cast<const uint4*>(g);
#endif
}

__device__ __forceinline__ void wait_async_copies() {
#if HAVE_ASYNC_LDS
#if __has_builtin(__builtin_amdgcn_s_wait_asynccnt)
    __builtin_amdgcn_s_wait_asynccnt(0);
#else
    asm volatile("s_wait_asynccnt 0x0" ::: "memory");
#endif
#endif
}

// ---------------------------- utility kernels ------------------------------

__global__ void zero_f32(float* __restrict__ p, long n) {
    long i = (long)blockIdx.x * blockDim.x + threadIdx.x;
    long stride = (long)gridDim.x * blockDim.x;
    for (; i < n; i += stride) p[i] = 0.0f;
}

// One 128-thread block per edge: gather x[src], scatter-add into agg[dst].
// agg is 100 MB -> resident in the 192 MB global L2, so the float atomics
// coalesce in-cache instead of hammering HBM.
__global__ void edge_scatter(const float* __restrict__ x,
                             const int*   __restrict__ ei,
                             const int*   __restrict__ ea,
                             float* __restrict__ agg,
                             float* __restrict__ cnt, int E) {
    int e = blockIdx.x;
    if (e >= E) return;
    if (ea[e] != 0) return;                 // R == 1: only relation 0
    int src = ei[e];
    int dst = ei[E + e];
    const float* xs = x   + (long)src * F_IN;
    float*       ad = agg + (long)dst * F_IN;
    for (int f = threadIdx.x; f < F_IN; f += blockDim.x)
        atomicAdd(&ad[f], xs[f]);
    if (threadIdx.x == 0) atomicAdd(&cnt[dst], 1.0f);
}

// x (f32, [N,500]) -> xbf (bf16, [N,512] zero-padded)
__global__ void cvt_x(const float* __restrict__ x, __bf16* __restrict__ xbf, int N) {
    long idx = (long)blockIdx.x * blockDim.x + threadIdx.x;
    long total = (long)N * KPAD;
    if (idx >= total) return;
    int i = (int)(idx / KPAD);
    int k = (int)(idx % KPAD);
    float v = (k < F_IN) ? x[(long)i * F_IN + k] : 0.0f;
    xbf[idx] = (__bf16)v;
}

// mean = agg / max(cnt,1) -> bf16 [N,512] zero-padded
__global__ void cvt_mean(const float* __restrict__ agg,
                         const float* __restrict__ cnt,
                         __bf16* __restrict__ mbf, int N) {
    long idx = (long)blockIdx.x * blockDim.x + threadIdx.x;
    long total = (long)N * KPAD;
    if (idx >= total) return;
    int i = (int)(idx / KPAD);
    int k = (int)(idx % KPAD);
    float v = 0.0f;
    if (k < F_IN) {
        float c = cnt[i];
        v = agg[(long)i * F_IN + k] * (1.0f / fmaxf(c, 1.0f));
    }
    mbf[idx] = (__bf16)v;
}

// W_root, W_rel (f32, [500,1024]) -> Wbf (bf16, [2,512,1024] zero-padded rows)
__global__ void cvt_w(const float* __restrict__ Wroot,
                      const float* __restrict__ Wrel,
                      __bf16* __restrict__ Wbf) {
    long idx = (long)blockIdx.x * blockDim.x + threadIdx.x;
    long total = 2L * KPAD * F_HID;
    if (idx >= total) return;
    int s   = (int)(idx / (KPAD * F_HID));
    int rem = (int)(idx % (KPAD * F_HID));
    int k = rem / F_HID;
    int n = rem % F_HID;
    const float* W = s ? Wrel : Wroot;
    float v = (k < F_IN) ? W[(long)k * F_HID + n] : 0.0f;
    Wbf[idx] = (__bf16)v;
}

// ------------------------- fused WMMA GEMM + pool --------------------------
// out[row,col] = bias[col] + x[row,:]@W_root + mean[row,:]@W_rel, ReLU,
// then atomicAdd into pooled[batch[row],col]. Block tile 128(M) x 64(N),
// 8 waves x (2x2) v_wmma_f32_16x16x32_bf16 fragments, double-buffered LDS
// fed by async global->LDS copies. The [N,1024] hidden activation is never
// materialized in memory.
__global__ __launch_bounds__(256) void
rgcn_gemm_pool(const __bf16* __restrict__ xbf,
               const __bf16* __restrict__ mbf,
               const __bf16* __restrict__ Wbf,
               const float*  __restrict__ bias,
               const int*    __restrict__ batch,
               float* __restrict__ pooled, int N) {
    // conflict-padded, 16B-aligned, double-buffered LDS tiles
    __shared__ __align__(16) __bf16 sA[2][128][40];  // 128 rows x 32 k
    __shared__ __align__(16) __bf16 sB[2][32][80];   // 32 k x 64 cols

    const int tid  = threadIdx.x;
    const int lane = tid & 31;
    const int wave = tid >> 5;
    const int wm   = wave >> 1;          // 0..3 : M sub-tile
    const int wn   = wave & 1;           // 0..1 : N sub-tile
    const int rowBase  = blockIdx.y * 128;
    const int colBase  = blockIdx.x * 64;
    const bool fullTile = (rowBase + 128) <= N;   // 390/391 blocks: no guards

    v8f acc[2][2] = {};

    constexpr int STEPS = 2 * (KPAD / 32);  // 2 sources x 16 K-tiles

    // stage one 128x32 A tile + 32x64 B tile for K-step `step` into buffer b
    auto stage = [&](int step, int b) {
        const int s  = step >> 4;              // 0: x/W_root  1: mean/W_rel
        const int k0 = (step & 15) << 5;
        const __bf16* A  = s ? mbf : xbf;
        const __bf16* Bp = Wbf + (long)s * KPAD * F_HID;
        // B tile: 256 x uint4, 1 per thread
        {
            const int kr = tid >> 3;
            const int nc = (tid & 7) << 3;
            async_copy_b128(Bp + (long)(k0 + kr) * F_HID + colBase + nc,
                            &sB[b][kr][nc]);
        }
        // A tile: 512 x uint4, 2 per thread
        if (fullTile) {
            const int r0 = tid >> 2, kc0 = (tid & 3) << 3;
            async_copy_b128(A + (long)(rowBase + r0) * KPAD + k0 + kc0,
                            &sA[b][r0][kc0]);
            const int c1 = tid + 256;
            const int r1 = c1 >> 2, kc1 = (c1 & 3) << 3;
            async_copy_b128(A + (long)(rowBase + r1) * KPAD + k0 + kc1,
                            &sA[b][r1][kc1]);
        } else {
            for (int c = tid; c < 512; c += 256) {
                const int r  = c >> 2;
                const int kc = (c & 3) << 3;
                const int gRow = rowBase + r;
                uint4 val = make_uint4(0u, 0u, 0u, 0u);
                if (gRow < N)
                    val = *reinterpret_cast<const uint4*>(
                        A + (long)gRow * KPAD + k0 + kc);
                *reinterpret_cast<uint4*>(&sA[b][r][kc]) = val;
            }
        }
    };

    stage(0, 0);
    for (int step = 0; step < STEPS; ++step) {
        const int buf = step & 1;
        wait_async_copies();       // our writes into sA/sB[buf] are done
        __syncthreads();           // everyone's writes are visible
        if (step + 1 < STEPS) stage(step + 1, buf ^ 1);   // overlap next tile

        // A fragment (16x32 bf16): lane<16 -> row M=lane, K 0-7 & 16-23;
        // lane>=16 -> row M=lane-16, K 8-15 & 24-31  (ISA 16-bit A layout)
        FragBF a[2], b[2];
        const int kOff = (lane < 16) ? 0 : 8;
        #pragma unroll
        for (int fr = 0; fr < 2; ++fr) {
            const int m = wm * 32 + fr * 16 + (lane & 15);
            a[fr].u[0] = *reinterpret_cast<const uint4*>(&sA[buf][m][kOff]);
            a[fr].u[1] = *reinterpret_cast<const uint4*>(&sA[buf][m][kOff + 16]);
        }
        // B fragment (32x16 bf16): col N=lane&15, K kOff..kOff+7 & kOff+16..kOff+23
        #pragma unroll
        for (int fc = 0; fc < 2; ++fc) {
            const int cb = wn * 32 + fc * 16 + (lane & 15);
            #pragma unroll
            for (int i = 0; i < 8; ++i) {
                b[fc].v[i]     = sB[buf][kOff + i][cb];
                b[fc].v[8 + i] = sB[buf][kOff + 16 + i][cb];
            }
        }
        #pragma unroll
        for (int fr = 0; fr < 2; ++fr)
            #pragma unroll
            for (int fc = 0; fc < 2; ++fc)
                acc[fr][fc] = __builtin_amdgcn_wmma_f32_16x16x32_bf16(
                    false, a[fr].v, false, b[fc].v,
                    (short)0, acc[fr][fc], false, false);
    }

    // epilogue: bias + ReLU + fused global_add_pool (atomics land in L2;
    // pooled is only 256 KB). C layout: VGPR r -> row r (lanes 0-15) or
    // row r+8 (lanes 16-31), col = lane & 15.
    #pragma unroll
    for (int fr = 0; fr < 2; ++fr) {
        #pragma unroll
        for (int fc = 0; fc < 2; ++fc) {
            const int col  = colBase + wn * 32 + fc * 16 + (lane & 15);
            const int rowb = rowBase + wm * 32 + fr * 16 + ((lane < 16) ? 0 : 8);
            const float bi = bias[col];
            #pragma unroll
            for (int r = 0; r < 8; ++r) {
                const int row = rowb + r;
                if (row < N) {
                    float v = acc[fr][fc][r] + bi;
                    v = v > 0.0f ? v : 0.0f;
                    const int g = batch[row];
                    atomicAdd(&pooled[(long)g * F_HID + col], v);
                }
            }
        }
    }
}

// pooled[64,1024] @ fc_W[1024,4] + fc_b -> out[64,4]  (tiny, one block)
__global__ void final_fc(const float* __restrict__ pooled,
                         const float* __restrict__ fcW,
                         const float* __restrict__ fcb,
                         float* __restrict__ out) {
    int t = threadIdx.x;
    if (t >= NGRAPH * 4) return;
    int g = t >> 2;
    int o = t & 3;
    float s = fcb[o];
    const float* pr = pooled + (long)g * F_HID;
    for (int k = 0; k < F_HID; ++k) s += pr[k] * fcW[k * 4 + o];
    out[t] = s;
}

// ------------------------------- launcher ----------------------------------

static inline size_t alignUp256(size_t v) { return (v + 255) & ~(size_t)255; }

extern "C" void kernel_launch(void* const* d_in, const int* in_sizes, int n_in,
                              void* d_out, int out_size, void* d_ws, size_t ws_size,
                              hipStream_t stream) {
    const float* x     = (const float*)d_in[0];
    const int*   ei    = (const int*)  d_in[1];
    const int*   ea    = (const int*)  d_in[2];
    const int*   batch = (const int*)  d_in[3];
    const float* Wrel  = (const float*)d_in[4];   // [1,500,1024]
    const float* Wroot = (const float*)d_in[5];   // [500,1024]
    const float* bias  = (const float*)d_in[6];
    const float* fcW   = (const float*)d_in[7];
    const float* fcb   = (const float*)d_in[8];

    const int N = in_sizes[0] / F_IN;
    const int E = in_sizes[1] / 2;

    // carve workspace
    char* p = (char*)d_ws;
    float*  agg    = (float*)p;  p += alignUp256((size_t)N * F_IN * 4);
    float*  cnt    = (float*)p;  p += alignUp256((size_t)N * 4);
    float*  pooled = (float*)p;  p += alignUp256((size_t)NGRAPH * F_HID * 4);
    __bf16* xbf    = (__bf16*)p; p += alignUp256((size_t)N * KPAD * 2);
    __bf16* mbf    = (__bf16*)p; p += alignUp256((size_t)N * KPAD * 2);
    __bf16* Wbf    = (__bf16*)p; p += alignUp256((size_t)2 * KPAD * F_HID * 2);
    (void)ws_size; (void)n_in; (void)out_size;

    // 1) zero accumulators
    hipLaunchKernelGGL(zero_f32, dim3(2048), dim3(256), 0, stream,
                       agg, (long)N * F_IN);
    hipLaunchKernelGGL(zero_f32, dim3(64), dim3(256), 0, stream, cnt, (long)N);
    hipLaunchKernelGGL(zero_f32, dim3(64), dim3(256), 0, stream,
                       pooled, (long)NGRAPH * F_HID);

    // 2) conversions independent of the scatter
    {
        long tot = 2L * KPAD * F_HID;
        hipLaunchKernelGGL(cvt_w, dim3((unsigned)((tot + 255) / 256)), dim3(256),
                           0, stream, Wroot, Wrel, Wbf);
    }
    {
        long tot = (long)N * KPAD;
        hipLaunchKernelGGL(cvt_x, dim3((unsigned)((tot + 255) / 256)), dim3(256),
                           0, stream, x, xbf, N);
    }

    // 3) edge gather/scatter-mean accumulation
    hipLaunchKernelGGL(edge_scatter, dim3(E), dim3(128), 0, stream,
                       x, ei, ea, agg, cnt, E);

    // 4) mean + bf16 convert
    {
        long tot = (long)N * KPAD;
        hipLaunchKernelGGL(cvt_mean, dim3((unsigned)((tot + 255) / 256)), dim3(256),
                           0, stream, agg, cnt, mbf, N);
    }

    // 5) fused WMMA GEMM + bias + ReLU + pooling
    hipLaunchKernelGGL(rgcn_gemm_pool,
                       dim3(F_HID / 64, (N + 127) / 128), dim3(256), 0, stream,
                       xbf, mbf, Wbf, bias, batch, pooled, N);

    // 6) tiny FC
    hipLaunchKernelGGL(final_fc, dim3(1), dim3(256), 0, stream,
                       pooled, fcW, fcb, (float*)d_out);
}
